// BitLinear_4372276707631
// MI455X (gfx1250) — compile-verified, matrix operand order checked by reading, not run on hardware
//
#include <hip/hip_runtime.h>
#include <hip/hip_bf16.h>

typedef __attribute__((ext_vector_type(16))) _Float16 v16h;
typedef __attribute__((ext_vector_type(8)))  _Float16 half8;
typedef __attribute__((ext_vector_type(4)))  _Float16 half4;
typedef __attribute__((ext_vector_type(8)))  float    v8f;

union FragA { v16h v; half8 h[2]; };

#define BM 128
#define BN 128
#define BK 32
#define LDP 40   // padded LDS row pitch in halfs (80B -> bank stride 20, minimal conflicts)

// ---------------------------------------------------------------------------
// CDNA5 async global->LDS (ASYNCcnt-tracked, no VGPR data round-trip)
// ---------------------------------------------------------------------------
__device__ __forceinline__ void async_load_b128(unsigned lds_off, const void* gptr) {
    asm volatile("global_load_async_to_lds_b128 %0, %1, off"
                 :: "v"(lds_off), "v"(gptr) : "memory");
}
__device__ __forceinline__ void wait_async0() {
    asm volatile("s_wait_asynccnt 0x0" ::: "memory");
}

// ---------------------------------------------------------------------------
// Block reduction (fixed-order tree -> deterministic)
// ---------------------------------------------------------------------------
__device__ __forceinline__ float blockReduceSum256(float v) {
    __shared__ float red[256];
    red[threadIdx.x] = v;
    __syncthreads();
    #pragma unroll
    for (int s = 128; s > 0; s >>= 1) {
        if ((int)threadIdx.x < s) red[threadIdx.x] += red[threadIdx.x + s];
        __syncthreads();
    }
    return red[0];
}

// ---------------------------------------------------------------------------
// 1) partial sums of |W|
// ---------------------------------------------------------------------------
__global__ __launch_bounds__(256) void reduce_abs_partial(const float* __restrict__ w,
                                                          int n, float* __restrict__ partials) {
    const float4* w4 = (const float4*)w;
    int n4 = n >> 2;
    int stride = gridDim.x * blockDim.x;
    float s = 0.f;
    for (int i = blockIdx.x * blockDim.x + threadIdx.x; i < n4; i += stride) {
        float4 v = w4[i];
        s += fabsf(v.x) + fabsf(v.y) + fabsf(v.z) + fabsf(v.w);
    }
    float t = blockReduceSum256(s);
    if (threadIdx.x == 0) partials[blockIdx.x] = t;
}

// ---------------------------------------------------------------------------
// 2) finalize scale = max(mean|W|, 1e-5)
// ---------------------------------------------------------------------------
__global__ __launch_bounds__(256) void reduce_finalize(const float* __restrict__ partials,
                                                       int np, int nelem,
                                                       float* __restrict__ scale) {
    float s = 0.f;
    for (int i = threadIdx.x; i < np; i += blockDim.x) s += partials[i];
    float t = blockReduceSum256(s);
    if (threadIdx.x == 0) *scale = fmaxf(t / (float)nelem, 1e-5f);
}

// ---------------------------------------------------------------------------
// 3) ternary quantize W -> fp16 ({-1,0,1} exact in fp16)
// ---------------------------------------------------------------------------
__global__ __launch_bounds__(256) void quantize_w(const float* __restrict__ w,
                                                  const float* __restrict__ scale_p,
                                                  _Float16* __restrict__ wq, int n) {
    int i = (blockIdx.x * blockDim.x + threadIdx.x) * 4;
    if (i >= n) return;
    float inv = 1.0f / (*scale_p);
    float4 v = *(const float4*)&w[i];
    half4 q;
    q.x = (_Float16)rintf(fminf(fmaxf(v.x * inv, -1.f), 1.f));
    q.y = (_Float16)rintf(fminf(fmaxf(v.y * inv, -1.f), 1.f));
    q.z = (_Float16)rintf(fminf(fmaxf(v.z * inv, -1.f), 1.f));
    q.w = (_Float16)rintf(fminf(fmaxf(v.w * inv, -1.f), 1.f));
    *(half4*)&wq[i] = q;
}

// ---------------------------------------------------------------------------
// 4) RMSNorm: one 256-thread block per row of length K
// ---------------------------------------------------------------------------
__global__ __launch_bounds__(256) void rmsnorm_rows(const float* __restrict__ x,
                                                    const float* __restrict__ gamma,
                                                    _Float16* __restrict__ xn, int K) {
    const int row = blockIdx.x;
    const float* xr = x + (size_t)row * K;
    _Float16* xo = xn + (size_t)row * K;

    float ss = 0.f;
    for (int c = threadIdx.x * 4; c < K; c += blockDim.x * 4) {
        float4 v = *(const float4*)&xr[c];
        ss += v.x * v.x + v.y * v.y + v.z * v.z + v.w * v.w;
    }
    float total = blockReduceSum256(ss);
    float inv = rsqrtf(total / (float)K + 1e-6f);

    for (int c = threadIdx.x * 4; c < K; c += blockDim.x * 4) {
        float4 v = *(const float4*)&xr[c];
        float4 g = *(const float4*)&gamma[c];
        half4 o;
        o.x = (_Float16)(v.x * inv * g.x);
        o.y = (_Float16)(v.y * inv * g.y);
        o.z = (_Float16)(v.z * inv * g.z);
        o.w = (_Float16)(v.w * inv * g.w);
        *(half4*)&xo[c] = o;
    }
}

// ---------------------------------------------------------------------------
// 5) WMMA GEMM: out[M,N] = xn[M,K] * wq[N,K]^T * scale
//    128x128 block tile, double-buffered async LDS staging, 8 waves,
//    each wave: 4x2 grid of 16x16 f32 accumulators.
// ---------------------------------------------------------------------------
__global__ __launch_bounds__(256) void bitlinear_gemm(const _Float16* __restrict__ A,
                                                      const _Float16* __restrict__ Bw,
                                                      const float* __restrict__ scale_p,
                                                      float* __restrict__ out,
                                                      int M, int N, int K) {
    __shared__ _Float16 sA[2][BM][LDP];
    __shared__ _Float16 sB[2][BN][LDP];

    const int tid   = threadIdx.x;
    const int lane  = tid & 31;
    const int wave  = tid >> 5;
    const int waveM = wave >> 2;          // 0..1 -> 64-row slab
    const int waveN = wave & 3;           // 0..3 -> 32-col slab

    const int m0 = blockIdx.y * BM;
    const int n0 = blockIdx.x * BN;

    // per-thread staging chunks: 512 chunks of 8 halfs per 128x32 tile
    const int ch0 = tid;
    const int ch1 = tid + 256;
    const int r0 = ch0 >> 2, c0 = (ch0 & 3) << 3;
    const int r1 = ch1 >> 2, c1 = (ch1 & 3) << 3;

    // LDS byte offsets: flat-addr low 32 bits == LDS offset (ISA aperture rule)
    const unsigned ldsA = (unsigned)(uintptr_t)&sA[0][0][0];
    const unsigned ldsB = (unsigned)(uintptr_t)&sB[0][0][0];
    const unsigned bufStride = (unsigned)(BM * LDP * sizeof(_Float16));
    const unsigned oA0 = (unsigned)(r0 * LDP + c0) * 2u;
    const unsigned oA1 = (unsigned)(r1 * LDP + c1) * 2u;

    const _Float16* Arow0 = A  + (size_t)(m0 + r0) * K + c0;
    const _Float16* Arow1 = A  + (size_t)(m0 + r1) * K + c1;
    const _Float16* Brow0 = Bw + (size_t)(n0 + r0) * K + c0;
    const _Float16* Brow1 = Bw + (size_t)(n0 + r1) * K + c1;

    v8f acc[4][2];
    #pragma unroll
    for (int mi = 0; mi < 4; ++mi)
        #pragma unroll
        for (int ni = 0; ni < 2; ++ni)
            acc[mi][ni] = (v8f){0.f, 0.f, 0.f, 0.f, 0.f, 0.f, 0.f, 0.f};

    const int row16 = lane & 15;
    const int klo   = (lane < 16) ? 0 : 8;   // ISA 16-bit A layout K-split

    const int nk = K / BK;

    // prologue: tile 0 -> buffer 0
    {
        unsigned bo = 0;
        async_load_b128(ldsA + bo + oA0, Arow0);
        async_load_b128(ldsA + bo + oA1, Arow1);
        async_load_b128(ldsB + bo + oA0, Brow0);
        async_load_b128(ldsB + bo + oA1, Brow1);
    }

    for (int i = 0; i < nk; ++i) {
        wait_async0();            // my tile-i async writes to LDS have landed
        __syncthreads();          // everyone's landed; everyone done reading buf[(i+1)&1]

        if (i + 1 < nk) {         // overlap tile i+1 DMA with tile i compute
            unsigned bo = (unsigned)((i + 1) & 1) * bufStride;
            size_t ko = (size_t)(i + 1) * BK;
            async_load_b128(ldsA + bo + oA0, Arow0 + ko);
            async_load_b128(ldsA + bo + oA1, Arow1 + ko);
            async_load_b128(ldsB + bo + oA0, Brow0 + ko);
            async_load_b128(ldsB + bo + oA1, Brow1 + ko);
        }

        const _Float16 (*cA)[LDP] = sA[i & 1];
        const _Float16 (*cB)[LDP] = sB[i & 1];

        FragA a[4], b[2];
        #pragma unroll
        for (int mi = 0; mi < 4; ++mi) {
            int r = waveM * 64 + mi * 16 + row16;
            a[mi].h[0] = *(const half8*)&cA[r][klo];
            a[mi].h[1] = *(const half8*)&cA[r][klo + 16];
        }
        #pragma unroll
        for (int ni = 0; ni < 2; ++ni) {
            int r = waveN * 32 + ni * 16 + row16;
            b[ni].h[0] = *(const half8*)&cB[r][klo];
            b[ni].h[1] = *(const half8*)&cB[r][klo + 16];
        }

        #pragma unroll
        for (int mi = 0; mi < 4; ++mi)
            #pragma unroll
            for (int ni = 0; ni < 2; ++ni)
                acc[mi][ni] = __builtin_amdgcn_wmma_f32_16x16x32_f16(
                    false, a[mi].v, false, b[ni].v,
                    (short)0, acc[mi][ni], false, false);
    }

    // ---- epilogue: D layout -> lane n = lane&15, row = vgpr + (lane>=16)*8 ----
    const float s  = *scale_p;
    const int hi8  = (lane >> 4) << 3;
    const int nn   = lane & 15;
    #pragma unroll
    for (int mi = 0; mi < 4; ++mi)
        #pragma unroll
        for (int ni = 0; ni < 2; ++ni) {
            size_t base = (size_t)(m0 + waveM * 64 + mi * 16 + hi8) * N
                        + (size_t)(n0 + waveN * 32 + ni * 16 + nn);
            #pragma unroll
            for (int r = 0; r < 8; ++r)
                out[base + (size_t)r * N] = acc[mi][ni][r] * s;
        }
}

// ---------------------------------------------------------------------------
// host launcher
// ---------------------------------------------------------------------------
extern "C" void kernel_launch(void* const* d_in, const int* in_sizes, int n_in,
                              void* d_out, int out_size, void* d_ws, size_t ws_size,
                              hipStream_t stream) {
    const float* x     = (const float*)d_in[0];
    const float* w     = (const float*)d_in[1];
    const float* gamma = (const float*)d_in[2];
    float* out = (float*)d_out;

    const int K = in_sizes[2];             // D_IN  (2048)
    const int M = in_sizes[0] / K;         // B*S   (8192)
    const int N = in_sizes[1] / K;         // D_OUT (8192)
    const int wsz = in_sizes[1];

    char* ws = (char*)d_ws;
    float*    partials = (float*)ws;                                     // 1024 f32
    float*    scale    = (float*)(ws + 4096);                            // 1 f32
    _Float16* wq       = (_Float16*)(ws + 8192);                         // N*K halfs
    _Float16* xn       = (_Float16*)(ws + 8192 + (size_t)N * K * 2);     // M*K halfs

    reduce_abs_partial<<<1024, 256, 0, stream>>>(w, wsz, partials);
    reduce_finalize  <<<1,    256, 0, stream>>>(partials, 1024, wsz, scale);
    quantize_w       <<<(wsz / 4 + 255) / 256, 256, 0, stream>>>(w, scale, wq, wsz);
    rmsnorm_rows     <<<M,    256, 0, stream>>>(x, gamma, xn, K);

    dim3 grid(N / BN, M / BM);
    bitlinear_gemm<<<grid, 256, 0, stream>>>(xn, wq, scale, out, M, N, K);
}
